// GlobalAttention_45028437131405
// MI455X (gfx1250) — matrix operand order repaired and checked by reading, not compiled, not run on hardware
//
#include <hip/hip_runtime.h>
#include <hip/hip_bf16.h>
#include <math.h>

typedef __bf16 bf16_t;
typedef __attribute__((ext_vector_type(16))) __bf16 bf16x16;
typedef __attribute__((ext_vector_type(8)))  __bf16 bf16x8;
typedef __attribute__((ext_vector_type(8)))  float  f32x8;

#define WMMA_BF16(a, b, c) \
  __builtin_amdgcn_wmma_f32_16x16x32_bf16(false, (a), false, (b), (short)0, (c), false, false)

constexpr int   kB    = 2;
constexpr int   kN    = 4096;
constexpr int   kC    = 256;   // D_MODEL
constexpr int   kH    = 8;
constexpr int   kD    = 32;    // HEAD_DIM
constexpr int   kRows = kB * kN;                       // 8192
// softmax scale folded together with log2(e): scores land in log2 domain,
// so p = exp2(s - m) is a single v_exp_f32.
constexpr float kScaleLog2 = 0.17677669529663687f * 1.4426950408889634f;

union AFrag { bf16x16 v; struct { bf16x8 lo, hi; } p; };

__device__ __forceinline__ bf16_t f2bf(float f) { return (bf16_t)f; }
__device__ __forceinline__ float  bf2f(bf16_t b) { return (float)b; }

__device__ __forceinline__ float fast_exp2(float x) {
#if __has_builtin(__builtin_amdgcn_exp2f)
  return __builtin_amdgcn_exp2f(x);
#else
  return exp2f(x);
#endif
}

// ---------------------------------------------------------------------------
// fp32 -> bf16 conversion (grid-stride)
// ---------------------------------------------------------------------------
__global__ void cvt_f32_bf16(const float* __restrict__ s, bf16_t* __restrict__ d, int n) {
  for (int i = blockIdx.x * blockDim.x + threadIdx.x; i < n; i += gridDim.x * blockDim.x)
    d[i] = f2bf(s[i]);
}

// ---------------------------------------------------------------------------
// Fused QKV projection: y = x @ W^T + b, WMMA bf16, f32 accum.
// grid = (kRows/16, 3), block = 128 (4 waves). Wave w computes cols [64w,64w+64).
// Q,K stored [b,h,n,d]; V stored transposed [b,h,d,n] for the P.V B-frags.
// ---------------------------------------------------------------------------
__global__ void __launch_bounds__(128)
qkv_gemm(const bf16_t* __restrict__ xb,
         const bf16_t* __restrict__ wq, const bf16_t* __restrict__ wk,
         const bf16_t* __restrict__ wv,
         const float* __restrict__ bq, const float* __restrict__ bk,
         const float* __restrict__ bv,
         bf16_t* __restrict__ Qd, bf16_t* __restrict__ Kd, bf16_t* __restrict__ Vt)
{
  const int lane = threadIdx.x & 31;
  const int wave = threadIdx.x >> 5;
  const int lm   = lane & 15;
  const int half = lane >> 4;
  const int row0 = blockIdx.x * 16;
  const int which = blockIdx.y;

  const bf16_t* W    = (which == 0) ? wq : (which == 1) ? wk : wv;
  const float*  bias = (which == 0) ? bq : (which == 1) ? bk : bv;

  f32x8 acc[4] = {};
  const bf16_t* arow = xb + (size_t)(row0 + lm) * kC + half * 16;

#pragma unroll
  for (int k = 0; k < kC; k += 32) {
    AFrag af;
    af.p.lo = *(const bf16x8*)(arow + k);
    af.p.hi = *(const bf16x8*)(arow + k + 8);
#pragma unroll
    for (int t = 0; t < 4; ++t) {
      const int col = wave * 64 + t * 16 + lm;
      bf16x16 bfr = *(const bf16x16*)(W + (size_t)col * kC + k + half * 16);
      acc[t] = WMMA_BF16(af.v, bfr, acc[t]);
    }
  }

#pragma unroll
  for (int t = 0; t < 4; ++t) {
    const int col = wave * 64 + t * 16 + lm;
    const float bc = bias[col];
    const int h = col >> 5, d = col & 31;
#pragma unroll
    for (int r = 0; r < 8; ++r) {
      const int gm = row0 + r + half * 8;        // global row over B*N
      const int b = gm >> 12, n = gm & (kN - 1);
      const bf16_t o = f2bf(acc[t][r] + bc);
      if (which == 2)
        Vt[(((size_t)b * kH + h) * kD + d) * kN + n] = o;
      else if (which == 0)
        Qd[(((size_t)b * kH + h) * kN + n) * kD + d] = o;
      else
        Kd[(((size_t)b * kH + h) * kN + n) * kD + d] = o;
    }
  }
}

// ---------------------------------------------------------------------------
// Flash attention: one wave per 16-query tile, online softmax over 4096 keys,
// 64 keys per iteration (4 score WMMAs + 4 P.V WMMAs, base-2 softmax).
// grid = B*H*(N/16)/4 = 1024, block = 128.
// ---------------------------------------------------------------------------
__global__ void __launch_bounds__(128)
flash_attn(const bf16_t* __restrict__ Qd, const bf16_t* __restrict__ Kd,
           const bf16_t* __restrict__ Vt, bf16_t* __restrict__ AO)
{
  // per-wave P tile (16x64 bf16), row stride 72 elem (144B): 16B aligned,
  // 36-dword bank rotation -> the 16 A-frag readers hit distinct banks.
  __shared__ __align__(16) bf16_t lds[4][16][72];

  const int lane = threadIdx.x & 31;
  const int wave = threadIdx.x >> 5;
  const int lm   = lane & 15;
  const int half = lane >> 4;
  const int tile = blockIdx.x * 4 + wave;   // over B*H*(N/16)
  const int bh = tile >> 8;                 // / (kN/16)
  const int mt = tile & 255;

  const bf16_t* Qb = Qd + (size_t)bh * kN * kD;
  const bf16_t* Kb = Kd + (size_t)bh * kN * kD;
  const bf16_t* Vb = Vt + (size_t)bh * kD * kN;

  AFrag qf;
  {
    const bf16_t* qrow = Qb + (size_t)(mt * 16 + lm) * kD + half * 16;
    qf.p.lo = *(const bf16x8*)qrow;
    qf.p.hi = *(const bf16x8*)(qrow + 8);
#pragma unroll
    for (int i = 0; i < 16; ++i)            // fold scale * log2(e) into Q
      qf.v[i] = f2bf(bf2f(qf.v[i]) * kScaleLog2);
  }

  f32x8 o0 = {}, o1 = {};
  float mrow[8], lrow[8];
#pragma unroll
  for (int r = 0; r < 8; ++r) { mrow[r] = -3.0e38f; lrow[r] = 0.0f; }

  for (int j0 = 0; j0 < kN; j0 += 64) {
    // prefetch the next 64-key block of K and V into cache (L2-resident data)
    if (j0 + 64 < kN) {
      __builtin_prefetch(Kb + (size_t)(j0 + 64 + lane) * kD, 0, 3);
      __builtin_prefetch(Vb + (size_t)lane * kN + j0 + 64, 0, 3);
    }

    // B-frags of K^T: lane = key j, contiguous 16 d-values per lane
    bf16x16 kb0 = *(const bf16x16*)(Kb + (size_t)(j0 +  0 + lm) * kD + half * 16);
    bf16x16 kb1 = *(const bf16x16*)(Kb + (size_t)(j0 + 16 + lm) * kD + half * 16);
    bf16x16 kb2 = *(const bf16x16*)(Kb + (size_t)(j0 + 32 + lm) * kD + half * 16);
    bf16x16 kb3 = *(const bf16x16*)(Kb + (size_t)(j0 + 48 + lm) * kD + half * 16);
    f32x8 z = {};
    f32x8 s0 = WMMA_BF16(qf.v, kb0, z);
    f32x8 s1 = WMMA_BF16(qf.v, kb1, z);
    f32x8 s2 = WMMA_BF16(qf.v, kb2, z);
    f32x8 s3 = WMMA_BF16(qf.v, kb3, z);

#pragma unroll
    for (int r = 0; r < 8; ++r) {
      // row max over 64 keys (local 4-way max, then 16-lane half-wave reduce)
      float vmax = fmaxf(fmaxf(s0[r], s1[r]), fmaxf(s2[r], s3[r]));
      vmax = fmaxf(vmax, __shfl_xor(vmax, 1, 32));
      vmax = fmaxf(vmax, __shfl_xor(vmax, 2, 32));
      vmax = fmaxf(vmax, __shfl_xor(vmax, 4, 32));
      vmax = fmaxf(vmax, __shfl_xor(vmax, 8, 32));
      const float mnew = fmaxf(mrow[r], vmax);
      const float corr = fast_exp2(mrow[r] - mnew);
      mrow[r] = mnew;
      const float p0 = fast_exp2(s0[r] - mnew);
      const float p1 = fast_exp2(s1[r] - mnew);
      const float p2 = fast_exp2(s2[r] - mnew);
      const float p3 = fast_exp2(s3[r] - mnew);
      float rs = (p0 + p1) + (p2 + p3);
      rs += __shfl_xor(rs, 1, 32);
      rs += __shfl_xor(rs, 2, 32);
      rs += __shfl_xor(rs, 4, 32);
      rs += __shfl_xor(rs, 8, 32);
      lrow[r] = lrow[r] * corr + rs;
      o0[r] *= corr;
      o1[r] *= corr;
      // stage P (C-layout -> LDS row-major 16x64 tile)
      const int prow = r + half * 8;
      lds[wave][prow][lm]      = f2bf(p0);
      lds[wave][prow][16 + lm] = f2bf(p1);
      lds[wave][prow][32 + lm] = f2bf(p2);
      lds[wave][prow][48 + lm] = f2bf(p3);
    }
    asm volatile("s_wait_dscnt 0" ::: "memory");

    // reload P as two A-fragments (K-chunks 0..31 and 32..63)
    AFrag pa0, pa1;
    pa0.p.lo = *(const bf16x8*)&lds[wave][lm][half * 16];
    pa0.p.hi = *(const bf16x8*)&lds[wave][lm][8 + half * 16];
    pa1.p.lo = *(const bf16x8*)&lds[wave][lm][32 + half * 16];
    pa1.p.hi = *(const bf16x8*)&lds[wave][lm][40 + half * 16];

    // B-frags of V: lane = d column, contiguous 16 j-values (V is transposed)
    bf16x16 vb00 = *(const bf16x16*)(Vb + (size_t)lm * kN        + j0 +  0 + half * 16);
    bf16x16 vb01 = *(const bf16x16*)(Vb + (size_t)lm * kN        + j0 + 32 + half * 16);
    bf16x16 vb10 = *(const bf16x16*)(Vb + (size_t)(16 + lm) * kN + j0 +  0 + half * 16);
    bf16x16 vb11 = *(const bf16x16*)(Vb + (size_t)(16 + lm) * kN + j0 + 32 + half * 16);
    o0 = WMMA_BF16(pa0.v, vb00, o0);
    o0 = WMMA_BF16(pa1.v, vb01, o0);
    o1 = WMMA_BF16(pa0.v, vb10, o1);
    o1 = WMMA_BF16(pa1.v, vb11, o1);
  }

  const int b = bh >> 3, h = bh & 7;
#pragma unroll
  for (int r = 0; r < 8; ++r) {
    const float inv = 1.0f / lrow[r];
    const int n = mt * 16 + r + half * 8;
    const size_t base = ((size_t)b * kN + n) * kC + h * kD;
    AO[base + lm]      = f2bf(o0[r] * inv);
    AO[base + 16 + lm] = f2bf(o1[r] * inv);
  }
}

// ---------------------------------------------------------------------------
// Output projection: Y = AO @ Wo^T + bo, fp32 output.
// grid = kRows/16, block = 128.
// ---------------------------------------------------------------------------
__global__ void __launch_bounds__(128)
out_proj(const bf16_t* __restrict__ A, const bf16_t* __restrict__ Wo,
         const float* __restrict__ bo, float* __restrict__ Y)
{
  const int lane = threadIdx.x & 31;
  const int wave = threadIdx.x >> 5;
  const int lm   = lane & 15;
  const int half = lane >> 4;
  const int row0 = blockIdx.x * 16;

  f32x8 acc[4] = {};
  const bf16_t* arow = A + (size_t)(row0 + lm) * kC + half * 16;

#pragma unroll
  for (int k = 0; k < kC; k += 32) {
    AFrag af;
    af.p.lo = *(const bf16x8*)(arow + k);
    af.p.hi = *(const bf16x8*)(arow + k + 8);
#pragma unroll
    for (int t = 0; t < 4; ++t) {
      const int col = wave * 64 + t * 16 + lm;
      bf16x16 bfr = *(const bf16x16*)(Wo + (size_t)col * kC + k + half * 16);
      acc[t] = WMMA_BF16(af.v, bfr, acc[t]);
    }
  }

#pragma unroll
  for (int t = 0; t < 4; ++t) {
    const int col = wave * 64 + t * 16 + lm;
    const float bc = bo[col];
#pragma unroll
    for (int r = 0; r < 8; ++r) {
      const int gm = row0 + r + half * 8;
      Y[(size_t)gm * kC + col] = acc[t][r] + bc;
    }
  }
}

// ---------------------------------------------------------------------------
extern "C" void kernel_launch(void* const* d_in, const int* in_sizes, int n_in,
                              void* d_out, int out_size, void* d_ws, size_t ws_size,
                              hipStream_t stream) {
  const float* x  = (const float*)d_in[0];
  const float* Wq = (const float*)d_in[1];
  const float* bq = (const float*)d_in[2];
  const float* Wk = (const float*)d_in[3];
  const float* bk = (const float*)d_in[4];
  const float* Wv = (const float*)d_in[5];
  const float* bv = (const float*)d_in[6];
  const float* Wo = (const float*)d_in[7];
  const float* bo = (const float*)d_in[8];

  // workspace partition (bf16 buffers)
  char* ws = (char*)d_ws;
  const size_t nXB = (size_t)kRows * kC;       // 2,097,152
  const size_t nW  = (size_t)kC * kC;          // 65,536
  bf16_t* xb  = (bf16_t*)ws;  ws += nXB * sizeof(bf16_t);
  bf16_t* wqb = (bf16_t*)ws;  ws += nW * sizeof(bf16_t);
  bf16_t* wkb = (bf16_t*)ws;  ws += nW * sizeof(bf16_t);
  bf16_t* wvb = (bf16_t*)ws;  ws += nW * sizeof(bf16_t);
  bf16_t* wob = (bf16_t*)ws;  ws += nW * sizeof(bf16_t);
  bf16_t* Qd  = (bf16_t*)ws;  ws += nXB * sizeof(bf16_t);
  bf16_t* Kd  = (bf16_t*)ws;  ws += nXB * sizeof(bf16_t);
  bf16_t* Vt  = (bf16_t*)ws;  ws += nXB * sizeof(bf16_t);
  bf16_t* AO  = (bf16_t*)ws;  ws += nXB * sizeof(bf16_t);

  cvt_f32_bf16<<<4096, 256, 0, stream>>>(x,  xb,  (int)nXB);
  cvt_f32_bf16<<<256,  256, 0, stream>>>(Wq, wqb, (int)nW);
  cvt_f32_bf16<<<256,  256, 0, stream>>>(Wk, wkb, (int)nW);
  cvt_f32_bf16<<<256,  256, 0, stream>>>(Wv, wvb, (int)nW);
  cvt_f32_bf16<<<256,  256, 0, stream>>>(Wo, wob, (int)nW);

  qkv_gemm<<<dim3(kRows / 16, 3), 128, 0, stream>>>(
      xb, wqb, wkb, wvb, bq, bk, bv, Qd, Kd, Vt);

  flash_attn<<<(kB * kH * (kN / 16)) / 4, 128, 0, stream>>>(Qd, Kd, Vt, AO);

  out_proj<<<kRows / 16, 128, 0, stream>>>(AO, wob, bo, (float*)d_out);
}
